// SwinTransformerBlock3D_70025146794393
// MI455X (gfx1250) — matrix operand order repaired and checked by reading, not compiled, not run on hardware
//
#include <hip/hip_runtime.h>

typedef _Float16 v8h  __attribute__((ext_vector_type(8)));
typedef _Float16 v16h __attribute__((ext_vector_type(16)));
typedef float    v8f  __attribute__((ext_vector_type(8)));

#define LANE (threadIdx.x & 31)
#define WAVE (threadIdx.x >> 5)

// ---- problem dims ----
#define ROWS   200704      // B*D*H*W tokens
#define NWIN   2048        // total windows
#define NTOK   98          // tokens per window
#define NTOKP  112         // padded to 7 tiles of 16
#define QSCALE 0.2041241452319315f   // 24^-0.5

// ---- workspace layout (bytes) ----
#define OFF_WQKV  0ull
#define OFF_WPROJ 55296ull
#define OFF_WFC1  73728ull
#define OFF_WFC2  147456ull
#define OFF_XLN   262144ull                      // xln f16 [2048][112][96]; reused by attn_out (same shape), ln2
#define SZ_XLN    44040192ull
#define OFF_Q     (OFF_XLN + SZ_XLN)             // q,k f16 [2048][4][112][32]; vT f16 [2048][4][32][128]
#define SZ_QK1    58720256ull
#define SZ_VT     67108864ull
#define OFF_K     (OFF_Q + SZ_QK1)
#define OFF_V     (OFF_K + SZ_QK1)
#define OFF_MLP1  OFF_Q                          // mlp1 (154.1 MB) reuses q/k/v (184.5 MB)
#define OFF_X2    (OFF_V + SZ_VT)                // x2 f32 [200704][96]

// ---------------- WMMA helpers ----------------
__device__ __forceinline__ v8f wmma_f16(v16h a, v16h b, v8f c) {
  return __builtin_amdgcn_wmma_f32_16x16x32_f16(false, a, false, b, (short)0, c, false, false);
}
// A fragment (16x32 f16): lane holds row m = row0+(l&15); halves 0..7 at k0+8*(l>>4), 8..15 at k0+16+8*(l>>4)
__device__ __forceinline__ v16h load_a_frag(const _Float16* p, int rstride, int row0, int k0) {
  int l = LANE;
  const _Float16* r = p + (size_t)(row0 + (l & 15)) * rstride + k0 + ((l >> 4) << 3);
  v8h lo = *(const v8h*)(r);
  v8h hi = *(const v8h*)(r + 16);
  return __builtin_shufflevector(lo, hi, 0,1,2,3,4,5,6,7,8,9,10,11,12,13,14,15);
}
// B fragment (32x16 f16, B supplied K-major i.e. [N][K]): lane holds col n = col0+(l&15);
// halves 0..7 at k0+8*(l>>4), 8..15 at k0+16+8*(l>>4)
__device__ __forceinline__ v16h load_b_frag(const _Float16* p, int rstride, int col0, int k0) {
  int l = LANE;
  const _Float16* r = p + (size_t)(col0 + (l & 15)) * rstride + k0 + ((l >> 4) << 3);
  v8h lo = *(const v8h*)(r);
  v8h hi = *(const v8h*)(r + 16);
  return __builtin_shufflevector(lo, hi, 0,1,2,3,4,5,6,7,8,9,10,11,12,13,14,15);
}

// ---------------- K0: weights -> f16 transposed [N][K] ----------------
__global__ void k_prep_weights(const float* __restrict__ qkv_w, const float* __restrict__ proj_w,
                               const float* __restrict__ fc1_w, const float* __restrict__ fc2_w,
                               _Float16* wqkvT, _Float16* wprojT, _Float16* wfc1T, _Float16* wfc2T) {
  int i = blockIdx.x * blockDim.x + threadIdx.x;
  if (i < 27648) { int n = i / 96, k = i % 96; wqkvT[i] = (_Float16)qkv_w[k * 288 + n]; return; }
  i -= 27648;
  if (i < 9216)  { int n = i / 96, k = i % 96; wprojT[i] = (_Float16)proj_w[k * 96 + n]; return; }
  i -= 9216;
  if (i < 36864) { int n = i / 96, k = i % 96; wfc1T[i] = (_Float16)fc1_w[k * 384 + n]; return; }
  i -= 36864;
  if (i < 36864) { int n = i / 384, k = i % 384; wfc2T[i] = (_Float16)fc2_w[k * 96 + n]; }
}

// ---------------- K1: zero fill (padded regions must be 0 each call) ----------------
__global__ void k_zero(float4* p, unsigned int n) {
  unsigned int i = blockIdx.x * blockDim.x + threadIdx.x;
  if (i < n) { float4 z; z.x = z.y = z.z = z.w = 0.f; p[i] = z; }
}

// ---------------- K2: LN1 + cyclic shift + window partition -> xln f16 ----------------
__global__ void __launch_bounds__(128) k_ln1_part(const float* __restrict__ x, const float* __restrict__ g,
                                                  const float* __restrict__ bb, _Float16* __restrict__ xln) {
  int widx = blockIdx.x;
  int token = threadIdx.x;
  if (token >= NTOK) return;
  int b = widx >> 8, wrem = widx & 255;
  int wd = wrem >> 6, wh = (wrem >> 3) & 7, ww = wrem & 7;
  int td = token / 49, th = (token % 49) / 7, tw = token % 7;
  int dsft = wd * 2 + td, hsft = wh * 7 + th, wsft = ww * 7 + tw;
  int d0 = (dsft + 1) & 7;                     // roll(-1) on D (size 8)
  int h0 = hsft + 3; if (h0 >= 56) h0 -= 56;   // roll(-3) on H
  int w0 = wsft + 3; if (w0 >= 56) w0 -= 56;   // roll(-3) on W
  const float* row = x + (((size_t)((b * 8 + d0) * 56 + h0)) * 56 + w0) * 96;
  float m = 0.f;
  for (int c = 0; c < 96; ++c) m += row[c];
  m *= (1.f / 96.f);
  float v = 0.f;
  for (int c = 0; c < 96; ++c) { float t = row[c] - m; v += t * t; }
  v *= (1.f / 96.f);
  float rs = rsqrtf(v + 1e-5f);
  _Float16* o = xln + ((size_t)widx * NTOKP + token) * 96;
  for (int c = 0; c < 96; ++c) o[c] = (_Float16)((row[c] - m) * rs * g[c] + bb[c]);
}

// ---------------- K3: QKV GEMM (M=2048*112, K=96, N=288) + split into padded Q/K/V^T ----------------
__global__ void __launch_bounds__(256) k_qkv(const _Float16* __restrict__ xln, const _Float16* __restrict__ wqkvT,
                                             const float* __restrict__ qkv_b, _Float16* __restrict__ qb,
                                             _Float16* __restrict__ kb, _Float16* __restrict__ vbT) {
  int mtile = blockIdx.x * 8 + WAVE;           // 14336 m-tiles
  int row0 = mtile * 16;
  int w = row0 / NTOKP;                        // wave-uniform: 16-row tile never crosses a window (112 = 7*16)
  int t0 = row0 % NTOKP;
  v16h a0 = load_a_frag(xln, 96, row0, 0);
  v16h a1 = load_a_frag(xln, 96, row0, 32);
  v16h a2 = load_a_frag(xln, 96, row0, 64);
  int l = LANE, nlo = l & 15, grp = l >> 4;
  int tb = t0 + grp * 8;
  for (int nt = 0; nt < 18; ++nt) {
    v16h b0 = load_b_frag(wqkvT, 96, nt * 16, 0);
    v16h b1 = load_b_frag(wqkvT, 96, nt * 16, 32);
    v16h b2 = load_b_frag(wqkvT, 96, nt * 16, 64);
    v8f acc = {};
    acc = wmma_f16(a0, b0, acc);
    acc = wmma_f16(a1, b1, acc);
    acc = wmma_f16(a2, b2, acc);
    int c = nt * 16 + nlo;
    int mat = c / 96, rem = c % 96, head = rem / 24, dd = rem % 24;   // mat is wave-uniform per nt
    float bias = qkv_b[c];
    if (mat == 2) {
      _Float16* dst = vbT + (((size_t)w * 4 + head) * 32 + dd) * 128 + tb;
#pragma unroll
      for (int r = 0; r < 8; ++r) dst[r] = (_Float16)(acc[r] + bias);
    } else {
      float scale = (mat == 0) ? QSCALE : 1.0f;
      _Float16* dst = ((mat == 0) ? qb : kb) + (((size_t)w * 4 + head) * NTOKP + tb) * 32 + dd;
#pragma unroll
      for (int r = 0; r < 8; ++r) dst[(size_t)r * 32] = (_Float16)((acc[r] + bias) * scale);
    }
  }
}

// ---------------- K4: windowed attention, one block per (window, head) ----------------
__global__ void __launch_bounds__(128) k_attn(const _Float16* __restrict__ qb, const _Float16* __restrict__ kb,
                                              const _Float16* __restrict__ vbT, const float* __restrict__ rpb,
                                              const float* __restrict__ maskp, _Float16* __restrict__ attn_out) {
  __shared__ float    sc[112 * 116];   // scores f32, stride 116 (bank-conflict padding)
  __shared__ _Float16 P [112 * 136];   // softmax probs f16, K padded to 128, stride 136
  int widx = blockIdx.x >> 2;
  int head = blockIdx.x & 3;
  const _Float16* qh = qb  + ((size_t)widx * 4 + head) * NTOKP * 32;
  const _Float16* kh = kb  + ((size_t)widx * 4 + head) * NTOKP * 32;
  const _Float16* vh = vbT + ((size_t)widx * 4 + head) * 32 * 128;   // [d][token], zero-padded to 128
  int l = LANE, nlo = l & 15, mloc = (l >> 4) * 8;
  // scores = Q * K^T  (49 tiles, K-dim = 32 in one WMMA), fused bias+mask in D-store
  for (int tile = WAVE; tile < 49; tile += 4) {
    int qt = tile / 7, kt = tile % 7;
    v16h aq = load_a_frag(qh, 32, qt * 16, 0);
    v16h bk = load_b_frag(kh, 32, kt * 16, 0);   // K row-major [token][d] == K^T in K-major
    v8f acc = {};
    acc = wmma_f16(aq, bk, acc);
    int nk = kt * 16 + nlo;
    int kd = nk / 49, kh2 = (nk % 49) / 7, kw = nk % 7;   // per-lane invariant across r
#pragma unroll
    for (int r = 0; r < 8; ++r) {
      int mq = qt * 16 + r + mloc;
      float s;
      if (mq < NTOK && nk < NTOK) {
        int qd = mq / 49, qh2 = (mq % 49) / 7, qw = mq % 7;
        int idx = (qd - kd + 1) * 169 + (qh2 - kh2 + 6) * 13 + (qw - kw + 6);
        s = acc[r] + rpb[idx * 4 + head] + maskp[((size_t)(widx & 255) * 98 + mq) * 98 + nk];
      } else {
        s = -1e30f;                               // padded keys/queries
      }
      sc[mq * 116 + nk] = s;
    }
  }
  __syncthreads();
  // row softmax (thread per row)
  int row = threadIdx.x;
  if (row < NTOKP) {
    float mx = -3.4e38f;
    for (int k = 0; k < NTOKP; ++k) mx = fmaxf(mx, sc[row * 116 + k]);
    float sum = 0.f;
    for (int k = 0; k < NTOKP; ++k) { float e = __expf(sc[row * 116 + k] - mx); sc[row * 116 + k] = e; sum += e; }
    float inv = 1.f / sum;
    for (int k = 0; k < NTOKP; ++k) P[row * 136 + k] = (_Float16)(sc[row * 116 + k] * inv);
    for (int k = NTOKP; k < 128; ++k) P[row * 136 + k] = (_Float16)0.f;
  }
  __syncthreads();
  // out = P * V (14 tiles, K-dim = 128 in 4 WMMA steps; batch loads, then chain WMMAs)
  for (int tile = WAVE; tile < 14; tile += 4) {
    int qt = tile / 2, dt = tile % 2;
    v16h ap[4], bv[4];
#pragma unroll
    for (int kt = 0; kt < 4; ++kt) {
      ap[kt] = load_a_frag(P, 136, qt * 16, kt * 32);
      bv[kt] = load_b_frag(vh, 128, dt * 16, kt * 32);
    }
    v8f acc = {};
#pragma unroll
    for (int kt = 0; kt < 4; ++kt) acc = wmma_f16(ap[kt], bv[kt], acc);
    int dcol = dt * 16 + nlo;
    if (dcol < 24) {
#pragma unroll
      for (int r = 0; r < 8; ++r) {
        int tok = qt * 16 + r + mloc;
        if (tok < NTOK)
          attn_out[((size_t)widx * NTOKP + tok) * 96 + head * 24 + dcol] = (_Float16)acc[r];
      }
    }
  }
}

// ---------------- K5: proj GEMM + window reverse + roll-back + residual -> x2 ----------------
__global__ void __launch_bounds__(256) k_proj(const _Float16* __restrict__ attn_out, const _Float16* __restrict__ wprojT,
                                              const float* __restrict__ proj_b, const float* __restrict__ x,
                                              float* __restrict__ x2) {
  int mtile = blockIdx.x * 8 + WAVE;           // 14336 m-tiles over padded 2048*112 rows
  int row0 = mtile * 16;
  int w = row0 / NTOKP;                        // wave-uniform
  int t0 = row0 % NTOKP;
  v16h a0 = load_a_frag(attn_out, 96, row0, 0);
  v16h a1 = load_a_frag(attn_out, 96, row0, 32);
  v16h a2 = load_a_frag(attn_out, 96, row0, 64);
  int l = LANE, nlo = l & 15, grp = l >> 4;
  // wave-uniform spatial decode of the window
  int b = w >> 8, wrem = w & 255;
  int wd = wrem >> 6, wh = (wrem >> 3) & 7, ww = wrem & 7;
  // per-r destination row base (hoisted out of the 6-tile N loop)
  size_t gibase[8];
  bool valid[8];
#pragma unroll
  for (int r = 0; r < 8; ++r) {
    int t = t0 + r + grp * 8;
    valid[r] = (t < NTOK);
    int tt = valid[r] ? t : 0;
    int td = tt / 49, th = (tt % 49) / 7, tw = tt % 7;
    int dsft = wd * 2 + td, hsft = wh * 7 + th, wsft = ww * 7 + tw;
    int d0 = (dsft + 1) & 7;
    int h0 = hsft + 3; if (h0 >= 56) h0 -= 56;
    int w0 = wsft + 3; if (w0 >= 56) w0 -= 56;
    gibase[r] = (((size_t)(b * 8 + d0) * 56 + h0) * 56 + w0) * 96;
  }
  for (int nt = 0; nt < 6; ++nt) {
    v16h b0 = load_b_frag(wprojT, 96, nt * 16, 0);
    v16h b1 = load_b_frag(wprojT, 96, nt * 16, 32);
    v16h b2 = load_b_frag(wprojT, 96, nt * 16, 64);
    v8f acc = {};
    acc = wmma_f16(a0, b0, acc);
    acc = wmma_f16(a1, b1, acc);
    acc = wmma_f16(a2, b2, acc);
    int c = nt * 16 + nlo;
    float pb = proj_b[c];
#pragma unroll
    for (int r = 0; r < 8; ++r) {
      if (valid[r]) {
        size_t gi = gibase[r] + c;
        x2[gi] = x[gi] + acc[r] + pb;
      }
    }
  }
}

// ---------------- K6: LN2 -> f16 ----------------
__global__ void __launch_bounds__(128) k_ln2(const float* __restrict__ x2, const float* __restrict__ g,
                                             const float* __restrict__ bb, _Float16* __restrict__ ln2) {
  int rowi = blockIdx.x * 128 + threadIdx.x;   // 1568 * 128 = 200704
  const float* row = x2 + (size_t)rowi * 96;
  float m = 0.f;
  for (int c = 0; c < 96; ++c) m += row[c];
  m *= (1.f / 96.f);
  float v = 0.f;
  for (int c = 0; c < 96; ++c) { float t = row[c] - m; v += t * t; }
  v *= (1.f / 96.f);
  float rs = rsqrtf(v + 1e-5f);
  _Float16* o = ln2 + (size_t)rowi * 96;
  for (int c = 0; c < 96; ++c) o[c] = (_Float16)((row[c] - m) * rs * g[c] + bb[c]);
}

// ---------------- K7: FC1 GEMM + exact GELU -> mlp1 f16 ----------------
__global__ void __launch_bounds__(256) k_fc1(const _Float16* __restrict__ ln2, const _Float16* __restrict__ wfc1T,
                                             const float* __restrict__ fc1_b, _Float16* __restrict__ mlp1) {
  int mtile = blockIdx.x * 8 + WAVE;
  int row0 = mtile * 16;
  v16h a0 = load_a_frag(ln2, 96, row0, 0);
  v16h a1 = load_a_frag(ln2, 96, row0, 32);
  v16h a2 = load_a_frag(ln2, 96, row0, 64);
  int l = LANE, nlo = l & 15, grp = l >> 4;
  for (int nt = 0; nt < 24; ++nt) {
    v16h b0 = load_b_frag(wfc1T, 96, nt * 16, 0);
    v16h b1 = load_b_frag(wfc1T, 96, nt * 16, 32);
    v16h b2 = load_b_frag(wfc1T, 96, nt * 16, 64);
    v8f acc = {};
    acc = wmma_f16(a0, b0, acc);
    acc = wmma_f16(a1, b1, acc);
    acc = wmma_f16(a2, b2, acc);
    int c = nt * 16 + nlo;
    float fb = fc1_b[c];
    _Float16* dst = mlp1 + (size_t)(row0 + grp * 8) * 384 + c;
#pragma unroll
    for (int r = 0; r < 8; ++r) {
      float xv = acc[r] + fb;
      float ge = 0.5f * xv * (1.f + erff(xv * 0.70710678118654752f));
      dst[(size_t)r * 384] = (_Float16)ge;
    }
  }
}

// ---------------- K8: FC2 GEMM + residual -> d_out f32 ----------------
__global__ void __launch_bounds__(256) k_fc2(const _Float16* __restrict__ mlp1, const _Float16* __restrict__ wfc2T,
                                             const float* __restrict__ fc2_b, const float* __restrict__ x2,
                                             float* __restrict__ out) {
  int mtile = blockIdx.x * 8 + WAVE;
  int row0 = mtile * 16;
  v16h af[12];
#pragma unroll
  for (int kt = 0; kt < 12; ++kt) af[kt] = load_a_frag(mlp1, 384, row0, kt * 32);
  int l = LANE, nlo = l & 15, grp = l >> 4;
  for (int nt = 0; nt < 6; ++nt) {
    v8f acc = {};
#pragma unroll
    for (int kc = 0; kc < 12; kc += 4) {       // batch 4 B loads, then 4 WMMAs
      v16h bf[4];
#pragma unroll
      for (int j = 0; j < 4; ++j) bf[j] = load_b_frag(wfc2T, 384, nt * 16, (kc + j) * 32);
#pragma unroll
      for (int j = 0; j < 4; ++j) acc = wmma_f16(af[kc + j], bf[j], acc);
    }
    int c = nt * 16 + nlo;
    float fb = fc2_b[c];
    size_t gi0 = (size_t)(row0 + grp * 8) * 96 + c;
#pragma unroll
    for (int r = 0; r < 8; ++r) {
      size_t gi = gi0 + (size_t)r * 96;
      out[gi] = x2[gi] + acc[r] + fb;
    }
  }
}

extern "C" void kernel_launch(void* const* d_in, const int* in_sizes, int n_in,
                              void* d_out, int out_size, void* d_ws, size_t ws_size,
                              hipStream_t stream) {
  const float* x      = (const float*)d_in[0];
  const float* maskp  = (const float*)d_in[1];
  const float* n1g    = (const float*)d_in[2];
  const float* n1b    = (const float*)d_in[3];
  const float* qkv_w  = (const float*)d_in[4];
  const float* qkv_b  = (const float*)d_in[5];
  const float* rpb    = (const float*)d_in[6];
  const float* proj_w = (const float*)d_in[7];
  const float* proj_b = (const float*)d_in[8];
  const float* n2g    = (const float*)d_in[9];
  const float* n2b    = (const float*)d_in[10];
  const float* fc1_w  = (const float*)d_in[11];
  const float* fc1_b  = (const float*)d_in[12];
  const float* fc2_w  = (const float*)d_in[13];
  const float* fc2_b  = (const float*)d_in[14];

  char* ws = (char*)d_ws;
  _Float16* wqkvT  = (_Float16*)(ws + OFF_WQKV);
  _Float16* wprojT = (_Float16*)(ws + OFF_WPROJ);
  _Float16* wfc1T  = (_Float16*)(ws + OFF_WFC1);
  _Float16* wfc2T  = (_Float16*)(ws + OFF_WFC2);
  _Float16* xln    = (_Float16*)(ws + OFF_XLN);
  _Float16* qb     = (_Float16*)(ws + OFF_Q);
  _Float16* kb     = (_Float16*)(ws + OFF_K);
  _Float16* vbT    = (_Float16*)(ws + OFF_V);
  _Float16* attn_o = (_Float16*)(ws + OFF_XLN);  // reuse (xln dead after QKV); padded [2048][112][96]
  _Float16* ln2    = (_Float16*)(ws + OFF_XLN);  // reuse (attn_out dead after proj)
  _Float16* mlp1   = (_Float16*)(ws + OFF_MLP1); // reuse (q/k/v dead after attention)
  float*    x2     = (float*)(ws + OFF_X2);
  float*    out    = (float*)d_out;

  // K0: weight convert/transpose (110592 elements)
  k_prep_weights<<<432, 256, 0, stream>>>(qkv_w, proj_w, fc1_w, fc2_w, wqkvT, wprojT, wfc1T, wfc2T);
  // K1: zero xln/attn_out (padded token rows) and q/k/vT (padded head-dim / token columns)
  k_zero<<<10752, 256, 0, stream>>>((float4*)(ws + OFF_XLN), (unsigned int)(SZ_XLN / 16));
  k_zero<<<45056, 256, 0, stream>>>((float4*)(ws + OFF_Q), (unsigned int)((2ull * SZ_QK1 + SZ_VT) / 16));
  // K2: LN1 + shift + partition
  k_ln1_part<<<NWIN, 128, 0, stream>>>(x, n1g, n1b, xln);
  // K3: QKV GEMM (14336 m-tiles / 8 waves per block)
  k_qkv<<<1792, 256, 0, stream>>>(xln, wqkvT, qkv_b, qb, kb, vbT);
  // K4: attention, one block per (window, head)
  k_attn<<<NWIN * 4, 128, 0, stream>>>(qb, kb, vbT, rpb, maskp, attn_o);
  // K5: proj + reverse + residual (padded rows: 14336 m-tiles)
  k_proj<<<1792, 256, 0, stream>>>(attn_o, wprojT, proj_b, x, x2);
  // K6: LN2
  k_ln2<<<1568, 128, 0, stream>>>(x2, n2g, n2b, ln2);
  // K7: FC1 + GELU
  k_fc1<<<1568, 256, 0, stream>>>(ln2, wfc1T, fc1_b, mlp1);
  // K8: FC2 + residual -> out
  k_fc2<<<1568, 256, 0, stream>>>(mlp1, wfc2T, fc2_b, x2, out);
}
